// ProteinStateReconstructor2D_41223096107484
// MI455X (gfx1250) — matrix operand order repaired and checked by reading, not compiled, not run on hardware
//
#include <hip/hip_runtime.h>
#include <hip/hip_bf16.h>
#include <math.h>

// ---------------------------------------------------------------------------
// Static problem dims (match the reference exactly)
// ---------------------------------------------------------------------------
constexpr int B_GRAPHS = 64;
constexpr int N_NODES  = 2048;
constexpr int NTOT     = B_GRAPHS * N_NODES;   // 131072 rows
constexpr int E_DIM    = 16;
constexpr int COND     = 64;
constexpr int POOLED   = 128;                  // 32*4
constexpr int HID      = 512;
constexpr float BN_EPS = 1e-5f;

typedef __attribute__((ext_vector_type(16))) _Float16 v16h;
typedef __attribute__((ext_vector_type(8)))  _Float16 v8h;
typedef __attribute__((ext_vector_type(8)))  float    v8f;

__device__ __forceinline__ float gelu_exact(float y) {
    return 0.5f * y * (1.0f + erff(y * 0.70710678118654752f));
}

// ---------------------------------------------------------------------------
// Small prep kernels
// ---------------------------------------------------------------------------
__global__ void zero_f32_kernel(float* __restrict__ p, int n) {
    int i = blockIdx.x * blockDim.x + threadIdx.x;
    if (i < n) p[i] = 0.0f;
}

// AdaptiveAvgPool2d over uniform bins: x[B*N, E] -> pooled[B, 128]
__global__ void pool_kernel(const float* __restrict__ x, float* __restrict__ pooled) {
    const int b = blockIdx.x;          // 64 blocks
    const int t = threadIdx.x;         // 128 threads: h = t/4, w = t%4
    const int h = t >> 2, w = t & 3;
    const float* xp = x + ((size_t)b * N_NODES + (size_t)h * 64) * E_DIM + w * 4;
    float s = 0.0f;
    #pragma unroll 4
    for (int r = 0; r < 64; ++r) {
        const float* rp = xp + (size_t)r * E_DIM;
        s += rp[0] + rp[1] + rp[2] + rp[3];
    }
    pooled[b * POOLED + t] = s * (1.0f / 256.0f);
}

// Per-graph bias: P1[b,n] = b1[n] + sum_k pooled[b,k] * W1[n, 64+k]
__global__ void p1_kernel(const float* __restrict__ pooled,
                          const float* __restrict__ W1,
                          const float* __restrict__ b1,
                          float* __restrict__ P1) {
    int idx = blockIdx.x * blockDim.x + threadIdx.x;   // 64*512
    int b = idx >> 9, n = idx & 511;
    const float* pp = pooled + b * POOLED;
    const float* wp = W1 + (size_t)n * (COND + POOLED) + COND;
    float s = b1[n];
    #pragma unroll 8
    for (int k = 0; k < POOLED; ++k) s = fmaf(pp[k], wp[k], s);
    P1[idx] = s;
}

__global__ void cvt_f16_kernel(const float* __restrict__ src,
                               _Float16* __restrict__ dst, int n) {
    int i = blockIdx.x * blockDim.x + threadIdx.x;
    if (i < n) dst[i] = (_Float16)src[i];
}

// W1c_h[n,k] = (f16) W1[n, k],  k < 64  (strided slice of the 192-wide W1)
__global__ void cvt_w1c_kernel(const float* __restrict__ W1,
                               _Float16* __restrict__ dst) {
    int i = blockIdx.x * blockDim.x + threadIdx.x;     // 512*64
    int n = i >> 6, k = i & 63;
    dst[i] = (_Float16)W1[(size_t)n * (COND + POOLED) + k];
}

// ---------------------------------------------------------------------------
// WMMA GEMM: D[NTOT,512] = act(A)[NTOT,K] @ Bw[512,K]^T + bias, f16 out,
// fused per-feature sum / sum-of-squares accumulation for BatchNorm stats.
// Block: 256 thr = 8 waves (2x4), wave tile 64x32, block tile 128x128.
// A fragments follow the 16-bit 16x32 A layout, B the 32x16 layout (wave32).
// All global offsets are 32-bit so loads use scalar-base + voffset form.
// ---------------------------------------------------------------------------
template<bool ACT, int K>
__global__ __launch_bounds__(256)
void gemm_bn_kernel(const _Float16* __restrict__ A,
                    const _Float16* __restrict__ Bw,        // [512, K] row-major (= W)
                    const float* __restrict__ biasVec,      // [512] or null
                    const float* __restrict__ biasPerGraph, // [64,512] or null
                    const float* __restrict__ actScale,     // [K] (ACT only)
                    const float* __restrict__ actShift,     // [K] (ACT only)
                    _Float16* __restrict__ D,               // [NTOT, 512]
                    float* __restrict__ statSum,            // [512]
                    float* __restrict__ statSS)             // [512]
{
    __shared__ float lsum[128];
    __shared__ float lss[128];
    if (threadIdx.x < 128) { lsum[threadIdx.x] = 0.0f; lss[threadIdx.x] = 0.0f; }
    __syncthreads();

    const int lane = threadIdx.x & 31;
    const int wave = threadIdx.x >> 5;
    const int wm   = wave & 1;           // 2 waves along M
    const int wn   = wave >> 1;          // 4 waves along N
    const int l16  = lane & 15;
    const int hs   = lane >> 4;          // half-wave select
    const int rowBase = blockIdx.x * 128 + wm * 64;
    const int colBase = blockIdx.y * 128 + wn * 32;

    v8f acc[4][2];
    #pragma unroll
    for (int mt = 0; mt < 4; ++mt)
        #pragma unroll
        for (int nt = 0; nt < 2; ++nt)
            #pragma unroll
            for (int v = 0; v < 8; ++v)
                acc[mt][nt][v] = 0.0f;

    // 32-bit base offsets (elements); everything fits in <2^31
    const unsigned aRow0 = (unsigned)rowBase * (unsigned)K + (unsigned)(hs * 8);
    const unsigned bCol0 = (unsigned)(colBase + l16) * (unsigned)K + (unsigned)(hs * 8);

    #pragma unroll 2
    for (int k0 = 0; k0 < K; k0 += 32) {
        // B fragments (32x16): lanes 0-15 hold K=k0..+7 and k0+16..+23 of col l16,
        // lanes 16-31 hold K=k0+8..+15 and k0+24..+31 (matches A).
        v16h bf[2];
        #pragma unroll
        for (int nt = 0; nt < 2; ++nt) {
            const unsigned boff = bCol0 + (unsigned)(nt * 16 * K) + (unsigned)k0;
            v8h b0 = *(const v8h*)(Bw + boff);
            v8h b1 = *(const v8h*)(Bw + boff + 16);
            #pragma unroll
            for (int i = 0; i < 8; ++i) { bf[nt][i] = b0[i]; bf[nt][i + 8] = b1[i]; }
        }
        // A fragments (16x32): lane<16 -> K=k0..+7 and k0+16..+23 of row l16;
        // lane>=16 -> K=k0+8..+15 and k0+24..+31. Two b128 loads per lane.
        v16h af[4];
        #pragma unroll
        for (int mt = 0; mt < 4; ++mt) {
            const unsigned aoff = aRow0 + (unsigned)((mt * 16 + l16) * K) + (unsigned)k0;
            v8h a0 = *(const v8h*)(A + aoff);
            v8h a1 = *(const v8h*)(A + aoff + 16);
            __builtin_prefetch(A + aoff + 32, 0, 3);   // speculative, WGP scope
            if (ACT) {
                const int kb = k0 + hs * 8;
                #pragma unroll
                for (int i = 0; i < 8; ++i) {
                    float y0 = fmaf((float)a0[i], actScale[kb + i],      actShift[kb + i]);
                    float y1 = fmaf((float)a1[i], actScale[kb + 16 + i], actShift[kb + 16 + i]);
                    af[mt][i]     = (_Float16)gelu_exact(y0);
                    af[mt][i + 8] = (_Float16)gelu_exact(y1);
                }
            } else {
                #pragma unroll
                for (int i = 0; i < 8; ++i) { af[mt][i] = a0[i]; af[mt][i + 8] = a1[i]; }
            }
        }
        #pragma unroll
        for (int mt = 0; mt < 4; ++mt)
            #pragma unroll
            for (int nt = 0; nt < 2; ++nt)
                acc[mt][nt] = __builtin_amdgcn_wmma_f32_16x16x32_f16(
                    false, af[mt], false, bf[nt], (short)0, acc[mt][nt], false, false);
    }

    // Epilogue: bias, f16 store (lanes 0-15 cover 16 contiguous cols of one
    // row -> 32B coalesced), BN stat partials via LDS atomics.
    const int graph = (blockIdx.x * 128) / N_NODES;   // uniform per block
    #pragma unroll
    for (int nt = 0; nt < 2; ++nt) {
        const int col = colBase + nt * 16 + l16;
        const float bias = biasPerGraph ? biasPerGraph[(unsigned)graph * HID + col]
                                        : biasVec[col];
        float s = 0.0f, ss = 0.0f;
        #pragma unroll
        for (int mt = 0; mt < 4; ++mt) {
            const unsigned row0 = (unsigned)(rowBase + mt * 16 + hs * 8);
            #pragma unroll
            for (int v = 0; v < 8; ++v) {
                float val = acc[mt][nt][v] + bias;
                D[(row0 + (unsigned)v) * (unsigned)HID + (unsigned)col] = (_Float16)val;
                s += val;
                ss = fmaf(val, val, ss);
            }
        }
        const int lc = wn * 32 + nt * 16 + l16;
        atomicAdd(&lsum[lc], s);     // ds_add_f32
        atomicAdd(&lss[lc],  ss);
    }
    __syncthreads();
    if (threadIdx.x < 128) {
        atomicAdd(statSum + blockIdx.y * 128 + threadIdx.x, lsum[threadIdx.x]);
        atomicAdd(statSS  + blockIdx.y * 128 + threadIdx.x, lss[threadIdx.x]);
    }
}

// Fold BN stats into per-feature affine: y = scale*x + shift
__global__ void finalize_bn_kernel(const float* __restrict__ sum,
                                   const float* __restrict__ ss,
                                   const float* __restrict__ g,
                                   const float* __restrict__ be,
                                   float* __restrict__ scale,
                                   float* __restrict__ shift) {
    int c = blockIdx.x * blockDim.x + threadIdx.x;
    if (c >= HID) return;
    const float inv = 1.0f / (float)NTOT;
    float m = sum[c] * inv;
    float v = fmaf(-m, m, ss[c] * inv);            // biased variance
    float s = g[c] * rsqrtf(v + BN_EPS);
    scale[c] = s;
    shift[c] = fmaf(-m, s, be[c]);
}

// Final layer: out[row, 0..2] = gelu(bn2(h2raw[row,:])) @ W3^T + b3
__global__ __launch_bounds__(256)
void layer3_kernel(const _Float16* __restrict__ h2,
                   const float* __restrict__ scale,
                   const float* __restrict__ shift,
                   const float* __restrict__ W3,   // [3, 512]
                   const float* __restrict__ b3,   // [3]
                   float* __restrict__ out) {      // [NTOT, 3]
    const unsigned row = blockIdx.x * blockDim.x + threadIdx.x;
    const _Float16* hp = h2 + row * (unsigned)HID;
    float a0 = 0.0f, a1 = 0.0f, a2 = 0.0f;
    for (int k = 0; k < HID; k += 8) {
        v8h hv = *(const v8h*)(hp + k);
        #pragma unroll
        for (int i = 0; i < 8; ++i) {
            float y = fmaf((float)hv[i], scale[k + i], shift[k + i]);
            float gl = gelu_exact(y);
            a0 = fmaf(gl, W3[k + i],           a0);
            a1 = fmaf(gl, W3[HID + k + i],     a1);
            a2 = fmaf(gl, W3[2 * HID + k + i], a2);
        }
    }
    out[(size_t)row * 3 + 0] = a0 + b3[0];
    out[(size_t)row * 3 + 1] = a1 + b3[1];
    out[(size_t)row * 3 + 2] = a2 + b3[2];
}

// ---------------------------------------------------------------------------
// Host launcher (graph-capture safe: only kernel launches on `stream`)
// ---------------------------------------------------------------------------
extern "C" void kernel_launch(void* const* d_in, const int* in_sizes, int n_in,
                              void* d_out, int out_size, void* d_ws, size_t ws_size,
                              hipStream_t stream) {
    (void)in_sizes; (void)n_in; (void)out_size; (void)ws_size;
    const float* x    = (const float*)d_in[0];
    /* d_in[1] = batch: unused — segments are uniform/sorted, graph = row / N */
    const float* cond = (const float*)d_in[2];
    const float* W1   = (const float*)d_in[3];
    const float* b1   = (const float*)d_in[4];
    const float* g1   = (const float*)d_in[5];
    const float* be1  = (const float*)d_in[6];
    const float* W2   = (const float*)d_in[7];
    const float* b2   = (const float*)d_in[8];
    const float* g2   = (const float*)d_in[9];
    const float* be2  = (const float*)d_in[10];
    const float* W3   = (const float*)d_in[11];
    const float* b3   = (const float*)d_in[12];
    float* out = (float*)d_out;

    // Workspace carve-out (256B aligned slices)
    char* ws = (char*)d_ws;
    size_t off = 0;
    auto carve = [&](size_t bytes) -> char* {
        char* p = ws + off;
        off += (bytes + 255) & ~(size_t)255;
        return p;
    };
    float*    pooled = (float*)carve((size_t)B_GRAPHS * POOLED * 4);
    float*    P1     = (float*)carve((size_t)B_GRAPHS * HID * 4);
    _Float16* W1c_h  = (_Float16*)carve((size_t)HID * COND * 2);
    _Float16* W2_h   = (_Float16*)carve((size_t)HID * HID * 2);
    float*    sum1   = (float*)carve(HID * 4);   // contiguous stats block
    float*    ss1    = (float*)carve(HID * 4);
    float*    sum2   = (float*)carve(HID * 4);
    float*    ss2    = (float*)carve(HID * 4);
    float*    scale1 = (float*)carve(HID * 4);
    float*    shift1 = (float*)carve(HID * 4);
    float*    scale2 = (float*)carve(HID * 4);
    float*    shift2 = (float*)carve(HID * 4);
    _Float16* A1     = (_Float16*)carve((size_t)NTOT * COND * 2);
    _Float16* h1raw  = (_Float16*)carve((size_t)NTOT * HID * 2);
    _Float16* h2raw  = (_Float16*)carve((size_t)NTOT * HID * 2);

    // 1) re-init stat accumulators every call (ws is not re-poisoned)
    zero_f32_kernel<<<(4 * HID + 255) / 256, 256, 0, stream>>>(sum1, 4 * HID);
    // 2) prep
    pool_kernel<<<B_GRAPHS, POOLED, 0, stream>>>(x, pooled);
    cvt_w1c_kernel<<<(HID * COND) / 256, 256, 0, stream>>>(W1, W1c_h);
    cvt_f16_kernel<<<(HID * HID) / 256, 256, 0, stream>>>(W2, W2_h, HID * HID);
    cvt_f16_kernel<<<(NTOT * COND) / 256, 256, 0, stream>>>(cond, A1, NTOT * COND);
    p1_kernel<<<(B_GRAPHS * HID) / 256, 256, 0, stream>>>(pooled, W1, b1, P1);
    // 3) layer 1: K=64 (pooled half folded into per-graph bias P1)
    gemm_bn_kernel<false, COND><<<dim3(NTOT / 128, HID / 128), 256, 0, stream>>>(
        A1, W1c_h, nullptr, P1, nullptr, nullptr, h1raw, sum1, ss1);
    finalize_bn_kernel<<<2, 256, 0, stream>>>(sum1, ss1, g1, be1, scale1, shift1);
    // 4) layer 2: K=512, BN1+GELU fused into A-fragment load
    gemm_bn_kernel<true, HID><<<dim3(NTOT / 128, HID / 128), 256, 0, stream>>>(
        h1raw, W2_h, b2, nullptr, scale1, shift1, h2raw, sum2, ss2);
    finalize_bn_kernel<<<2, 256, 0, stream>>>(sum2, ss2, g2, be2, scale2, shift2);
    // 5) layer 3 with BN2+GELU fused
    layer3_kernel<<<NTOT / 256, 256, 0, stream>>>(h2raw, scale2, shift2, W3, b3, out);
}
